// BaseAttention_4587025072513
// MI455X (gfx1250) — matrix operand order repaired and checked by reading, not compiled, not run on hardware
//
#include <hip/hip_runtime.h>
#include <hip/hip_bf16.h>
#include <math.h>

// ---------------------------------------------------------------------------
// MI455X (gfx1250) attention: f16 WMMA everywhere, flash-style single pass.
//   B=4, S=4096, H=512.  q/k stored f16 row-major, v stored f16 TRANSPOSED.
//   K tiles staged into LDS with the Tensor Data Mover (tensor_load_to_lds).
// ---------------------------------------------------------------------------

typedef __attribute__((ext_vector_type(16))) _Float16     v16h;
typedef __attribute__((ext_vector_type(8)))  _Float16     v8h;
typedef __attribute__((ext_vector_type(8)))  float        v8f;
typedef __attribute__((ext_vector_type(4)))  unsigned int v4u;
typedef __attribute__((ext_vector_type(8)))  int          v8i;
typedef __attribute__((ext_vector_type(4)))  int          v4i;

#if __has_builtin(__builtin_amdgcn_tensor_load_to_lds)
#define USE_TDM 1
#endif

#if __has_builtin(__builtin_amdgcn_s_wait_tensorcnt)
#define WAIT_TENSOR() __builtin_amdgcn_s_wait_tensorcnt(0)
#else
#define WAIT_TENSOR() asm volatile("s_wait_tensorcnt 0x0" ::: "memory")
#endif

__device__ __forceinline__ v16h cat16(v8h lo, v8h hi) {
    return __builtin_shufflevector(lo, hi, 0,1,2,3,4,5,6,7,8,9,10,11,12,13,14,15);
}

__device__ __forceinline__ v8f wmma_f16(v16h a, v16h b, v8f c) {
    // v_wmma_f32_16x16x32_f16: (neg_a, A, neg_b, B, c_mod, C, reuse_a, reuse_b)
    return __builtin_amdgcn_wmma_f32_16x16x32_f16(false, a, false, b, (short)0, c,
                                                  false, false);
}

// exchange value with lane^16 inside the wave32 (ds_swizzle SWAPX16)
__device__ __forceinline__ float swap16(float x) {
    int i = __float_as_int(x);
    i = __builtin_amdgcn_ds_swizzle(i, 0x401f);  // xor=0x10, and=0x1f, or=0
    return __int_as_float(i);
}

// ---------------------------------------------------------------------------
// f32 -> f16 conversion (vector-of-8 per thread)
// ---------------------------------------------------------------------------
__global__ __launch_bounds__(256) void cvt_f32_f16(const float* __restrict__ in,
                                                   _Float16* __restrict__ out, int n) {
    int i = (blockIdx.x * 256 + threadIdx.x) * 8;
    if (i + 7 < n) {
        v8f f = *(const v8f*)(in + i);
        v8h h;
        #pragma unroll
        for (int j = 0; j < 8; ++j) h[j] = (_Float16)f[j];
        *(v8h*)(out + i) = h;
    }
}

// ---------------------------------------------------------------------------
// Projection GEMM:  Y[m][o] = sum_h X[m][h] * W[o][h]      (y = x @ W.T)
// M = B*S = 16384, N = K = 512.  Block: 8 waves, each wave = 16(M) x 64(N).
// transposeStore==0: Y row-major f16 [M][512]
// transposeStore==1: Yt[b][h][s] f16 (for V), b = m>>12, s = m&4095
// ---------------------------------------------------------------------------
__global__ __launch_bounds__(256) void gemm_proj(const _Float16* __restrict__ X,
                                                 const _Float16* __restrict__ W,
                                                 _Float16* __restrict__ Y,
                                                 int transposeStore) {
    constexpr int H = 512, S = 4096;
    const int wave = threadIdx.x >> 5;
    const int lane = threadIdx.x & 31;
    const int l16  = lane & 15;
    const int hh   = lane >> 4;            // 0 for lanes 0-15, 1 for 16-31
    const int m0   = blockIdx.x * 128 + wave * 16;
    const int n0   = blockIdx.y * 64;

    const _Float16* arow = X + (size_t)(m0 + l16) * H + hh * 8;
    v8f acc[4] = {};

    #pragma unroll 4
    for (int kk = 0; kk < H; kk += 32) {
        // A fragment (16x32): lane row m0+l16; chunks at k = kk+8*hh and +16
        v16h afrag = cat16(*(const v8h*)(arow + kk),
                           *(const v8h*)(arow + kk + 16));
        #pragma unroll
        for (int nt = 0; nt < 4; ++nt) {
            // B fragment (32x16): B[k][n] = W[n][k]; lane col n0+nt*16+l16,
            // chunks at k = kk + 8*hh and +16
            const _Float16* bp = W + (size_t)(n0 + nt * 16 + l16) * H + kk + hh * 8;
            v16h bfrag = cat16(*(const v8h*)bp, *(const v8h*)(bp + 16));
            acc[nt] = wmma_f16(afrag, bfrag, acc[nt]);
        }
    }

    const int rbase = hh * 8;              // C layout: vgpr r -> row rbase+r
    if (!transposeStore) {
        #pragma unroll
        for (int nt = 0; nt < 4; ++nt) {
            const int o = n0 + nt * 16 + l16;
            #pragma unroll
            for (int r = 0; r < 8; ++r)
                Y[(size_t)(m0 + rbase + r) * H + o] = (_Float16)acc[nt][r];
        }
    } else {
        const int b = m0 >> 12;
        const int sbase = (m0 & (S - 1)) + rbase;   // rows contiguous in r
        #pragma unroll
        for (int nt = 0; nt < 4; ++nt) {
            const int h = n0 + nt * 16 + l16;
            v8h hv;
            #pragma unroll
            for (int r = 0; r < 8; ++r) hv[r] = (_Float16)acc[nt][r];
            *(v8h*)(Y + ((size_t)(b * H + h) << 12) + sbase) = hv;
        }
    }
}

// ---------------------------------------------------------------------------
// Flash attention.  One workgroup (4 waves) per (batch, 16-query tile).
// Wave w owns output columns [w*128, w*128+128).
// The 32-key K tile (32x512 f16 = 32 KB) is DMA'd into LDS once per tile by
// the Tensor Data Mover and shared by all 4 waves (4x less vmem traffic).
// Scores computed transposed (rows = keys, cols = queries) so softmax stats
// are per-lane scalars; O accumulated transposed so the final store is
// contiguous.
// ---------------------------------------------------------------------------
__global__ __launch_bounds__(128) void attn_flash(const _Float16* __restrict__ Q,
                                                  const _Float16* __restrict__ K,
                                                  const _Float16* __restrict__ Vt,
                                                  float* __restrict__ O,
                                                  float scale) {
    constexpr int S = 4096, H = 512;
    __shared__ _Float16 Ks[32 * H];       // 32 KB: current K tile (keys x H)
    __shared__ _Float16 Ps[4][16 * 32];   //  4 KB: per-wave P tile (f16)

    const int b  = blockIdx.x >> 8;       // 256 query tiles per batch
    const int qs = (blockIdx.x & 255) * 16;

    const int wave = threadIdx.x >> 5;
    const int lane = threadIdx.x & 31;
    const int l16  = lane & 15;
    const int hh   = lane >> 4;
    const int h0   = wave * 128;

    const _Float16* Kg = K  + (size_t)b * S * H;
    const _Float16* Vg = Vt + (size_t)b * H * S;
    _Float16* Pw = &Ps[wave][0];

    float m_run = -1e30f, l_run = 0.0f;
    v8f oacc[8] = {};                     // O^T: 8 subtiles of 16 h-cols

    // B-frag base for Q^T (direct from global; tile is L0-resident)
    const _Float16* qrow = Q + ((size_t)b * S + qs + l16) * H + hh * 8;

#ifdef USE_TDM
    // Tensor DMA descriptor (2D): tensor 4096x512 f16, tile 32x512.
    const unsigned int ldsoff = (unsigned int)(unsigned long long)(uintptr_t)Ks;
    const unsigned long long kbase = (unsigned long long)(uintptr_t)Kg;
    const v8i g1 = { 0x00010000,          // data_size = 1 (2 bytes), mask=0
                     (int)(512u << 16),   // tensor_dim0 = 512 (low 16 in [31:16])
                     (int)(4096u << 16),  // tensor_dim1 = 4096
                     (int)(512u << 16),   // tile_dim0 = 512
                     32,                  // tile_dim1 = 32
                     512,                 // tensor_dim0_stride = 512
                     0, 0 };
    const v4i gz = { 0, 0, 0, 0 };
#endif

    for (int kt = 0; kt < S; kt += 32) {
        __syncthreads();                  // all waves done with previous Ks
#ifdef USE_TDM
        if (wave == 0) {
            const unsigned long long ga = kbase + (unsigned long long)kt * (H * 2);
            v4u g0;
            g0[0] = 1u;                                      // count = 1 (valid)
            g0[1] = ldsoff;                                  // LDS byte address
            g0[2] = (unsigned int)ga;                        // global addr lo
            g0[3] = (unsigned int)((ga >> 32) & 0x01ffffffu) // global addr hi
                    | (2u << 30);                            // type = 2 (image)
#if defined(__clang_major__) && __clang_major__ >= 23
            const v8i z8 = { 0, 0, 0, 0, 0, 0, 0, 0 };
            __builtin_amdgcn_tensor_load_to_lds(g0, g1, gz, gz, z8, 0);
#else
            __builtin_amdgcn_tensor_load_to_lds(g0, g1, gz, gz, 0);
#endif
            WAIT_TENSOR();
        }
#else
        // fallback: cooperative copy (K tile is contiguous in global)
        {
            const _Float16* src = Kg + (size_t)kt * H;
            for (int i = threadIdx.x * 8; i < 32 * H; i += 128 * 8)
                *(v8h*)(Ks + i) = *(const v8h*)(src + i);
        }
#endif
        __syncthreads();                  // Ks ready for all waves

        __builtin_prefetch(Vg + (size_t)(h0 + l16) * S + kt + 32, 0, 0);

        // ---- scores S^T (32 keys x 16 queries), K-dim = H -----------------
        const _Float16* k0 = Ks + (size_t)l16 * H + hh * 8;   // keys 0-15
        const _Float16* k1 = k0 + 16 * H;                     // keys 16-31

        v8f c0 = {}, c1 = {};
        #pragma unroll 4
        for (int hk = 0; hk < H; hk += 32) {
            v16h bq = cat16(*(const v8h*)(qrow + hk),
                            *(const v8h*)(qrow + hk + 16));
            v16h a0 = cat16(*(const v8h*)(k0 + hk), *(const v8h*)(k0 + hk + 16));
            v16h a1 = cat16(*(const v8h*)(k1 + hk), *(const v8h*)(k1 + hk + 16));
            c0 = wmma_f16(a0, bq, c0);
            c1 = wmma_f16(a1, bq, c1);
        }

        // ---- online softmax (query row = l16, per-lane scalar stats) ------
        float tmax = -1e30f;
        #pragma unroll
        for (int r = 0; r < 8; ++r) {
            c0[r] *= scale; c1[r] *= scale;
            tmax = fmaxf(tmax, fmaxf(c0[r], c1[r]));
        }
        tmax = fmaxf(tmax, swap16(tmax));            // merge the two half-waves
        const float m_new = fmaxf(m_run, tmax);
        const float corr  = __expf(m_run - m_new);

        v8h p0, p1;
        float rsum = 0.0f;
        #pragma unroll
        for (int r = 0; r < 8; ++r) {
            float e0 = __expf(c0[r] - m_new);
            float e1 = __expf(c1[r] - m_new);
            p0[r] = (_Float16)e0; p1[r] = (_Float16)e1;
            rsum += e0 + e1;
        }
        rsum  = rsum + swap16(rsum);
        l_run = l_run * corr + rsum;
        m_run = m_new;

        // ---- stash P[qrow][key] in LDS (single b128 each) -----------------
        asm volatile("" ::: "memory");               // keep LDS ops ordered
        *(v8h*)(Pw + l16 * 32 + hh * 8)      = p0;   // keys  hh*8 .. +7
        *(v8h*)(Pw + l16 * 32 + hh * 8 + 16) = p1;   // keys 16+hh*8 .. +7
        asm volatile("" ::: "memory");

        // B frag = P^T: lane col = l16, keys hh*8..+7 and 16+hh*8..+7
        v16h pfrag = cat16(*(const v8h*)(Pw + l16 * 32 + hh * 8),
                           *(const v8h*)(Pw + l16 * 32 + hh * 8 + 16));

        // ---- O^T += Vt_chunk * P^T  (A contiguous from transposed V) ------
        const _Float16* vbase = Vg + (size_t)(h0 + l16) * S + kt + hh * 8;
        #pragma unroll
        for (int nt = 0; nt < 8; ++nt) {
            const _Float16* vp = vbase + (size_t)(nt * 16) * S;
            v16h vfrag = cat16(*(const v8h*)vp, *(const v8h*)(vp + 16));
            v8f cc = oacc[nt];
            #pragma unroll
            for (int r = 0; r < 8; ++r) cc[r] *= corr;
            oacc[nt] = wmma_f16(vfrag, pfrag, cc);
        }
    }

    // ---- epilogue: divide by l and store (8 contiguous f32 per lane) ------
    const float linv = 1.0f / l_run;
    float* Orow = O + ((size_t)b * S + qs + l16) * H;
    #pragma unroll
    for (int nt = 0; nt < 8; ++nt) {
        const int hc = h0 + nt * 16 + hh * 8;        // r -> hc+r contiguous
        v8f ov;
        #pragma unroll
        for (int r = 0; r < 8; ++r) ov[r] = oacc[nt][r] * linv;
        *(v8f*)(Orow + hc) = ov;
    }
}

// ---------------------------------------------------------------------------
extern "C" void kernel_launch(void* const* d_in, const int* in_sizes, int n_in,
                              void* d_out, int out_size, void* d_ws, size_t ws_size,
                              hipStream_t stream) {
    (void)in_sizes; (void)n_in; (void)out_size; (void)ws_size;
    constexpr int B = 4, S = 4096, H = 512;
    constexpr size_t M = (size_t)B * S;

    const float* x  = (const float*)d_in[0];
    const float* Wq = (const float*)d_in[1];
    const float* Wk = (const float*)d_in[2];
    const float* Wv = (const float*)d_in[3];
    float* out = (float*)d_out;

    _Float16* xh  = (_Float16*)d_ws;           // M*H
    _Float16* wqh = xh  + M * H;               // H*H
    _Float16* wkh = wqh + (size_t)H * H;
    _Float16* wvh = wkh + (size_t)H * H;
    _Float16* qh  = wvh + (size_t)H * H;       // M*H  (row-major)
    _Float16* kh  = qh  + M * H;               // M*H  (row-major)
    _Float16* vt  = kh  + M * H;               // B*H*S (transposed)

    // f32 -> f16 conversions
    cvt_f32_f16<<<(int)(M * H / (256 * 8)), 256, 0, stream>>>(x,  xh,  (int)(M * H));
    cvt_f32_f16<<<(H * H) / (256 * 8),      256, 0, stream>>>(Wq, wqh, H * H);
    cvt_f32_f16<<<(H * H) / (256 * 8),      256, 0, stream>>>(Wk, wkh, H * H);
    cvt_f32_f16<<<(H * H) / (256 * 8),      256, 0, stream>>>(Wv, wvh, H * H);

    // QKV projections (WMMA)
    dim3 gg((unsigned)(M / 128), H / 64);
    gemm_proj<<<gg, 256, 0, stream>>>(xh, wqh, qh, 0);
    gemm_proj<<<gg, 256, 0, stream>>>(xh, wkh, kh, 0);
    gemm_proj<<<gg, 256, 0, stream>>>(xh, wvh, vt, 1);   // V stored transposed

    // flash attention: one workgroup per (batch, 16-query tile)
    attn_flash<<<B * (S / 16), 128, 0, stream>>>(qh, kh, vt, out,
                                                 1.0f / sqrtf((float)H));
}
